// DeepGraphConvLayer_77962246357247
// MI455X (gfx1250) — matrix operand order repaired and checked by reading, not compiled, not run on hardware
//
#include <hip/hip_runtime.h>
#include <hip/hip_bf16.h>
#include <math.h>

// ---------------- problem constants (fixed by the reference) ----------------
#define BGRAPH 512
#define NODE   128
#define EPG    1024
#define HEADS  8
#define DFEAT  32
#define INF_   256          // input feature dim
#define COLS   256          // H*D
#define NROWS  (BGRAPH*NODE)      // 65536
#define ETOT   (BGRAPH*EPG)       // 524288
#define NEG_SLOPE 0.2f
#define BN_EPS 1e-5f

typedef __attribute__((ext_vector_type(16))) __bf16 v16bf;
typedef __attribute__((ext_vector_type(8)))  float  v8f;

// ---------------- workspace layout (bytes) ----------------
// feat   : NROWS*COLS f32      = 67,108,864
// p      : ETOT*HEADS f32      = 16,777,216   (logits, then exp(logit-max))
// mx     : NROWS*HEADS f32     =  2,097,152
// ssum   : NROWS*HEADS f32     =  2,097,152
// wbf    : COLS*INF_  bf16     =    131,072
// sums   : COLS f32, sumsq : COLS f32
#define WS_FEAT  0
#define WS_P     (WS_FEAT + (size_t)NROWS*COLS*4)
#define WS_MX    (WS_P    + (size_t)ETOT*HEADS*4)
#define WS_SSUM  (WS_MX   + (size_t)NROWS*HEADS*4)
#define WS_WBF   (WS_SSUM + (size_t)NROWS*HEADS*4)
#define WS_SUMS  (WS_WBF  + (size_t)COLS*INF_*2)
#define WS_SUMSQ (WS_SUMS + (size_t)COLS*4)

// ---------------- helpers ----------------
__device__ __forceinline__ void atomicMaxFloat(float* addr, float val) {
  if (!signbit(val)) atomicMax((int*)addr, __float_as_int(val));
  else               atomicMin((unsigned int*)addr, __float_as_uint(val));
}

// ---------------- 0: convert fc_w to bf16, init reductions ----------------
__global__ __launch_bounds__(256) void cvt_w_kernel(const float* __restrict__ w,
                                                    __bf16* __restrict__ wbf) {
  int i = blockIdx.x * 256 + threadIdx.x;
  if (i < COLS * INF_) wbf[i] = (__bf16)w[i];
}

__global__ __launch_bounds__(256) void init_kernel(float* mx, float* ssum,
                                                   float* sums, float* sumsq) {
  int i = blockIdx.x * 256 + threadIdx.x;
  int nm = NROWS * HEADS;
  for (int k = i; k < nm; k += gridDim.x * 256) { mx[k] = -INFINITY; ssum[k] = 0.f; }
  if (i < COLS) { sums[i] = 0.f; sumsq[i] = 0.f; }
}

// ---------------- 1: feat = x @ fc_w^T + fc_b  (bf16 WMMA, f32 accum) ------
// grid: NROWS/64 = 1024 blocks, 256 threads = 8 waves.
// wave w: rows = blk*64 + (w>>1)*16 .. +16,  cols = (w&1)*128 .. +128
__global__ __launch_bounds__(256)
void gemm_feat_kernel(const float* __restrict__ x, const __bf16* __restrict__ wbf,
                      const float* __restrict__ bias, float* __restrict__ feat) {
  const int lane = threadIdx.x & 31;
  const int wave = threadIdx.x >> 5;
  const int half = lane >> 4;      // lane half selects K sub-offset 16*half (contiguous k order)
  const int l15  = lane & 15;
  const int rowBase = blockIdx.x * 64 + (wave >> 1) * 16;
  const int colBase = (wave & 1) * 128;

  v8f acc[8];
  #pragma unroll
  for (int nt = 0; nt < 8; ++nt) {
    float b = bias[colBase + nt * 16 + l15];   // C/D: lane -> N, bias indep of M
    #pragma unroll
    for (int i = 0; i < 8; ++i) acc[nt][i] = b;
  }

  const float* xrow = x + (size_t)(rowBase + l15) * INF_;  // A: lane -> M

  #pragma unroll
  for (int ks = 0; ks < 8; ++ks) {           // K = 256 in steps of 32
    // A fragment 16x32 bf16: elems 0-15 -> K = k0..k0+15 (k0 = ks*32 + 16*half)
    const int k0 = ks * 32 + half * 16;
    v16bf afrag;
    #pragma unroll
    for (int i = 0; i < 8; ++i) {
      afrag[i]     = (__bf16)xrow[k0 + i];
      afrag[i + 8] = (__bf16)xrow[k0 + 8 + i];
    }
    const int kb = ks * 32 + half * 16;      // B: lane-half selects K+16
    #pragma unroll
    for (int nt = 0; nt < 8; ++nt) {
      const int col = colBase + nt * 16 + l15;          // B: lane -> N
      // feat = x @ w^T  =>  B[k][n] = w[n][k] : contiguous along k in row n
      v16bf bfrag = *(const v16bf*)(wbf + (size_t)col * INF_ + kb);
      acc[nt] = __builtin_amdgcn_wmma_f32_16x16x32_bf16(
          false, afrag, false, bfrag, (short)0, acc[nt], false, false);
    }
  }

  #pragma unroll
  for (int nt = 0; nt < 8; ++nt) {
    const int col = colBase + nt * 16 + l15;
    #pragma unroll
    for (int r = 0; r < 8; ++r)              // D: VGPR r -> M = r + 8*half
      feat[(size_t)(rowBase + r + 8 * half) * COLS + col] = acc[nt][r];
  }
}

// ---------------- 2: edge logits + segment max --------------------------
// thread per (edge, head); idx = e*8 + h
__global__ __launch_bounds__(256)
void edge_logits_kernel(const float* __restrict__ feat, const float* __restrict__ attn_w,
                        const int* __restrict__ src, const int* __restrict__ dst,
                        float* __restrict__ logits, float* __restrict__ mx) {
  size_t idx = (size_t)blockIdx.x * 256 + threadIdx.x;
  if (idx >= (size_t)ETOT * HEADS) return;
  int h = (int)(idx & 7);
  size_t e = idx >> 3;
  int sN = src[e], dN = dst[e];
  const float4* fs = (const float4*)(feat + (size_t)sN * COLS + h * DFEAT);
  const float4* fd = (const float4*)(feat + (size_t)dN * COLS + h * DFEAT);
  const float4* aw = (const float4*)(attn_w + h * DFEAT);
  float acc = 0.f;
  #pragma unroll
  for (int i = 0; i < 8; ++i) {
    float4 a = fs[i], b = fd[i], w = aw[i];
    float v0 = a.x + b.x, v1 = a.y + b.y, v2 = a.z + b.z, v3 = a.w + b.w;
    v0 = v0 > 0.f ? v0 : v0 * NEG_SLOPE;
    v1 = v1 > 0.f ? v1 : v1 * NEG_SLOPE;
    v2 = v2 > 0.f ? v2 : v2 * NEG_SLOPE;
    v3 = v3 > 0.f ? v3 : v3 * NEG_SLOPE;
    acc += v0 * w.x + v1 * w.y + v2 * w.z + v3 * w.w;
  }
  logits[idx] = acc;
  atomicMaxFloat(&mx[(size_t)dN * HEADS + h], acc);
}

// ---------------- 3: p = exp(logit - mx[dst]); segment sum ---------------
__global__ __launch_bounds__(256)
void edge_exp_kernel(float* __restrict__ p, const float* __restrict__ mx,
                     const int* __restrict__ dst, float* __restrict__ ssum) {
  size_t idx = (size_t)blockIdx.x * 256 + threadIdx.x;
  if (idx >= (size_t)ETOT * HEADS) return;
  int h = (int)(idx & 7);
  size_t e = idx >> 3;
  int dN = dst[e];
  float v = __expf(p[idx] - mx[(size_t)dN * HEADS + h]);
  p[idx] = v;
  atomicAdd(&ssum[(size_t)dN * HEADS + h], v);
}

// ---------------- 4: attn output [B, H, EPG] -----------------------------
__global__ __launch_bounds__(256)
void attn_out_kernel(const float* __restrict__ p, const float* __restrict__ ssum,
                     const int* __restrict__ dst, float* __restrict__ attn) {
  size_t idx = (size_t)blockIdx.x * 256 + threadIdx.x;
  if (idx >= (size_t)ETOT * HEADS) return;
  int h = (int)(idx & 7);
  size_t e = idx >> 3;
  int b = (int)(e >> 10), el = (int)(e & 1023);
  float a = p[idx] / ssum[(size_t)dst[e] * HEADS + h];
  attn[((size_t)b * HEADS + h) * EPG + el] = a;
}

// ---------------- 5: message aggregation in LDS + residual ---------------
// block per (graph, head-half): 64KB LDS tile of 128 nodes x 128 cols
__global__ __launch_bounds__(256)
void aggregate_kernel(const float* __restrict__ feat, const float* __restrict__ p,
                      const float* __restrict__ ssum, const int* __restrict__ src,
                      const int* __restrict__ dst, const float* __restrict__ x,
                      float* __restrict__ out) {
  __shared__ float smem[NODE * 128];
  const int g  = blockIdx.x >> 1;
  const int hh = blockIdx.x & 1;                  // column half: cols hh*128..+128
  for (int i = threadIdx.x; i < NODE * 128; i += 256) smem[i] = 0.f;
  __syncthreads();

  const int c  = threadIdx.x & 127;               // column within half
  const int eg = threadIdx.x >> 7;                // 2 edge lanes
  const int h  = hh * 4 + (c >> 5);               // head of this column
  const size_t ebase = (size_t)g * EPG;
  for (int e = eg; e < EPG; e += 2) {
    size_t eG = ebase + e;
    int sN = src[eG], dN = dst[eG];
    float a = p[eG * HEADS + h] / ssum[(size_t)dN * HEADS + h];
    float v = a * feat[(size_t)sN * COLS + hh * 128 + c];
    atomicAdd(&smem[(dN & (NODE - 1)) * 128 + c], v);   // ds_add_f32
  }
  __syncthreads();

  for (int i = threadIdx.x; i < NODE * 128; i += 256) {
    int nl = i >> 7, cc = i & 127;
    size_t node = (size_t)g * NODE + nl;
    out[node * COLS + hh * 128 + cc] = smem[i] + x[node * COLS + hh * 128 + cc];
  }
}

// ---------------- 6: BN channel stats ------------------------------------
__global__ __launch_bounds__(256)
void bn_stats_kernel(const float* __restrict__ out, float* __restrict__ sums,
                     float* __restrict__ sumsq) {
  int c = threadIdx.x;                       // channel
  size_t row0 = (size_t)blockIdx.x * 128;
  float s = 0.f, s2 = 0.f;
  for (int r = 0; r < 128; ++r) {
    float v = out[(row0 + r) * COLS + c];
    s += v; s2 += v * v;
  }
  atomicAdd(&sums[c], s);
  atomicAdd(&sumsq[c], s2);
}

// ---------------- 7: BN apply + ReLU (in place) --------------------------
__global__ __launch_bounds__(256)
void bn_apply_kernel(float* __restrict__ out, const float* __restrict__ sums,
                     const float* __restrict__ sumsq, const float* __restrict__ gamma,
                     const float* __restrict__ beta) {
  size_t idx = (size_t)blockIdx.x * 256 + threadIdx.x;
  if (idx >= (size_t)NROWS * COLS) return;
  int c = (int)(idx & (COLS - 1));
  const float invN = 1.f / (float)NROWS;
  float mean = sums[c] * invN;
  float var  = sumsq[c] * invN - mean * mean;
  float v = (out[idx] - mean) * rsqrtf(var + BN_EPS) * gamma[c] + beta[c];
  out[idx] = v > 0.f ? v : 0.f;
}

// ---------------- launch ----------------
extern "C" void kernel_launch(void* const* d_in, const int* in_sizes, int n_in,
                              void* d_out, int out_size, void* d_ws, size_t ws_size,
                              hipStream_t stream) {
  const float* x      = (const float*)d_in[0];
  const float* fc_w   = (const float*)d_in[1];
  const float* fc_b   = (const float*)d_in[2];
  const float* attn_w = (const float*)d_in[3];
  const float* gamma  = (const float*)d_in[4];
  const float* beta   = (const float*)d_in[5];
  const int*   src    = (const int*)d_in[6];
  const int*   dst    = (const int*)d_in[7];

  char* ws = (char*)d_ws;
  float*  feat  = (float*)(ws + WS_FEAT);
  float*  p     = (float*)(ws + WS_P);
  float*  mx    = (float*)(ws + WS_MX);
  float*  ssum  = (float*)(ws + WS_SSUM);
  __bf16* wbf   = (__bf16*)(ws + WS_WBF);
  float*  sums  = (float*)(ws + WS_SUMS);
  float*  sumsq = (float*)(ws + WS_SUMSQ);

  float* out  = (float*)d_out;                       // [NROWS, 256]
  float* attn = out + (size_t)NROWS * COLS;          // [B, H, EPG]

  cvt_w_kernel<<<(COLS * INF_ + 255) / 256, 256, 0, stream>>>(fc_w, wbf);
  init_kernel<<<2048, 256, 0, stream>>>(mx, ssum, sums, sumsq);

  gemm_feat_kernel<<<NROWS / 64, 256, 0, stream>>>(x, wbf, fc_b, feat);

  const int eh_blocks = (int)(((size_t)ETOT * HEADS + 255) / 256);
  edge_logits_kernel<<<eh_blocks, 256, 0, stream>>>(feat, attn_w, src, dst, p, mx);
  edge_exp_kernel<<<eh_blocks, 256, 0, stream>>>(p, mx, dst, ssum);
  attn_out_kernel<<<eh_blocks, 256, 0, stream>>>(p, ssum, dst, attn);

  aggregate_kernel<<<BGRAPH * 2, 256, 0, stream>>>(feat, p, ssum, src, dst, x, out);

  bn_stats_kernel<<<NROWS / 128, 256, 0, stream>>>(out, sums, sumsq);
  bn_apply_kernel<<<(int)(((size_t)NROWS * COLS + 255) / 256), 256, 0, stream>>>(
      out, sums, sumsq, gamma, beta);
}
